// xLSTMModel_54485955117299
// MI455X (gfx1250) — matrix operand order repaired and checked by reading, not compiled, not run on hardware
//
#include <hip/hip_runtime.h>
#include <cstdint>
#include <cstddef>

// ---------------------------------------------------------------------------
// Types for CDNA5 WMMA (wave32): v_wmma_f32_16x16x32_bf16
// ---------------------------------------------------------------------------
typedef __attribute__((ext_vector_type(8)))  float   v8f;
typedef __attribute__((ext_vector_type(16))) __bf16  v16bf;
typedef __attribute__((ext_vector_type(8)))  __bf16  v8bf;

union Frag { v16bf v; v8bf h[2]; };

__device__ __forceinline__ v8bf ld8(const __bf16* p) { return *(const v8bf*)p; }

__device__ __forceinline__ Frag zero_frag() {
  Frag f;
#pragma unroll
  for (int i = 0; i < 16; ++i) f.v[i] = (__bf16)0.0f;
  return f;
}

enum { EPI_NONE = 0, EPI_RES = 1, EPI_CAP = 2, EPI_GATE = 3 };

// ---- fragment loaders (ISA 16-bit A 16x32 / B 32x16 VGPR layouts) ----------
__device__ __forceinline__ void load_a(Frag (&fa)[2], const __bf16* __restrict__ A,
                                       int lda, int m0, int l15, int hi, int k0) {
#pragma unroll
  for (int mt = 0; mt < 2; ++mt) {
    const __bf16* p = A + (size_t)(m0 + mt * 16 + l15) * lda + k0 + hi * 8;
    fa[mt].h[0] = ld8(p);
    fa[mt].h[1] = ld8(p + 16);
  }
}

__device__ __forceinline__ void load_b(Frag (&fb)[4], const __bf16* __restrict__ B,
                                       int ldb, int n0, int l15, int hi, int k0) {
#pragma unroll
  for (int nt = 0; nt < 4; ++nt) {
    const __bf16* p = B + (size_t)(n0 + nt * 16 + l15) * ldb + k0 + hi * 8;
    fb[nt].h[0] = ld8(p);
    fb[nt].h[1] = ld8(p + 16);
  }
}

__device__ __forceinline__ void mma8(v8f (&acc)[2][4], const Frag (&fa)[2],
                                     const Frag (&fb)[4]) {
#pragma unroll
  for (int mt = 0; mt < 2; ++mt)
#pragma unroll
    for (int nt = 0; nt < 4; ++nt)
      acc[mt][nt] = __builtin_amdgcn_wmma_f32_16x16x32_bf16(
          false, fa[mt].v, false, fb[nt].v, (short)0, acc[mt][nt], false, false);
}

// ---------------------------------------------------------------------------
// Generic GEMM: C[M,N] = A[M,K] (bf16, row-major, lda) * B[N,K]^T (bf16, ldb)
// K is a multiple of 64 (buffers zero-padded by the converters).
// Per wave: 2x4 tiles of 16x16 => 32(M) x 64(N). Block = 8 waves => N=512.
// Fast path: full tile in range -> branch-free double-buffered k-loop with
// peeled tail and clamped global_prefetch of the streamed B rows.
// Edge path: per-fragment guards (ragged N blocks, tiny gate GEMMs).
// ---------------------------------------------------------------------------
__global__ __launch_bounds__(256) void k_gemm(
    const __bf16* __restrict__ A, int lda,
    const __bf16* __restrict__ B, int ldb,
    float* __restrict__ C, int ldc,
    const float* __restrict__ Cadd,
    int M, int N, int K, int epi, float cap)
{
  const int lane = threadIdx.x & 31;
  const int wave = threadIdx.x >> 5;
  const int hi   = (lane >> 4) & 1;
  const int l15  = lane & 15;
  const int m0 = blockIdx.y * 32;
  const int n0 = (blockIdx.x * 8 + wave) * 64;
  if (n0 >= N || m0 >= M) return;

  v8f acc[2][4];
#pragma unroll
  for (int mt = 0; mt < 2; ++mt)
#pragma unroll
    for (int nt = 0; nt < 4; ++nt) { v8f zz = {}; acc[mt][nt] = zz; }

  if ((m0 + 32 <= M) && (n0 + 64 <= N)) {
    // ------- fast path: branch-free double-buffered loop, peeled tail ------
    Frag a0[2], b0[4], a1[2], b1[4];
    load_a(a0, A, lda, m0, l15, hi, 0);
    load_b(b0, B, ldb, n0, l15, hi, 0);
    int k0 = 0;
    for (; k0 + 64 < K; k0 += 64) {
      const int pk = (k0 + 512 < K) ? (k0 + 512) : (K - 64);
#pragma unroll
      for (int nt = 0; nt < 4; ++nt)
        __builtin_prefetch(B + (size_t)(n0 + nt * 16 + l15) * ldb + pk, 0, 3);
      load_a(a1, A, lda, m0, l15, hi, k0 + 32);
      load_b(b1, B, ldb, n0, l15, hi, k0 + 32);
      mma8(acc, a0, b0);
      load_a(a0, A, lda, m0, l15, hi, k0 + 64);
      load_b(b0, B, ldb, n0, l15, hi, k0 + 64);
      mma8(acc, a1, b1);
    }
    // tail: chunks at k0 and k0+32 (K multiple of 64 guarantees both exist)
    load_a(a1, A, lda, m0, l15, hi, k0 + 32);
    load_b(b1, B, ldb, n0, l15, hi, k0 + 32);
    mma8(acc, a0, b0);
    mma8(acc, a1, b1);
  } else {
    // ---------------- edge path: guarded fragments -------------------------
    const Frag fz = zero_frag();
    for (int k0 = 0; k0 < K; k0 += 32) {
      Frag fa[2], fb[4];
#pragma unroll
      for (int mt = 0; mt < 2; ++mt) {
        const int row = m0 + mt * 16 + l15;
        if (row < M) {
          const __bf16* p = A + (size_t)row * lda + k0 + hi * 8;
          fa[mt].h[0] = ld8(p);
          fa[mt].h[1] = ld8(p + 16);
        } else fa[mt] = fz;
      }
#pragma unroll
      for (int nt = 0; nt < 4; ++nt) {
        const int col = n0 + nt * 16 + l15;
        if (col < N) {
          const __bf16* p = B + (size_t)col * ldb + k0 + hi * 8;
          fb[nt].h[0] = ld8(p);
          fb[nt].h[1] = ld8(p + 16);
        } else fb[nt] = fz;
      }
      mma8(acc, fa, fb);
    }
  }

  // C/D layout: lanes 0-15 -> M rows 0..7 in vgprs, lanes 16-31 -> rows 8..15
#pragma unroll
  for (int mt = 0; mt < 2; ++mt) {
    const int rbase = m0 + mt * 16 + hi * 8;
#pragma unroll
    for (int nt = 0; nt < 4; ++nt) {
      const int col = n0 + nt * 16 + l15;
      if (col >= N) continue;
#pragma unroll
      for (int r8 = 0; r8 < 8; ++r8) {
        const int row = rbase + r8;
        if (row >= M) continue;
        float val = acc[mt][nt][r8];
        if (epi == EPI_RES)       val += Cadd[(size_t)row * ldc + col];
        else if (epi == EPI_CAP)  val = cap * tanhf(val / cap);
        else if (epi == EPI_GATE) { float t = 15.0f * tanhf(val * (1.0f / 15.0f));
                                    val = 1.0f / (1.0f + expf(-t)); }
        C[(size_t)row * ldc + col] = val;
      }
    }
  }
}

// ---------------------------------------------------------------------------
// Causal depth-4 conv as 4 shifted GEMMs fused into one kernel.
// A = xn bf16 [M=B*S, K], W = bf16 [4][N][K] (per-tap, transposed to [out,in]).
// out[r,:] = sum_t xn[r+t-3,:] @ W[t]^T   (only if (r%S)+t-3 >= 0)
// Fast path requires (m0 % S) != 0 so all taps are in range for the block.
// ---------------------------------------------------------------------------
__global__ __launch_bounds__(256) void k_conv_gemm(
    const __bf16* __restrict__ A, int lda,
    const __bf16* __restrict__ W, int ldb,
    float* __restrict__ C, int ldc,
    int M, int N, int K, int S)
{
  const int lane = threadIdx.x & 31;
  const int wave = threadIdx.x >> 5;
  const int hi   = (lane >> 4) & 1;
  const int l15  = lane & 15;
  const int m0 = blockIdx.y * 32;
  const int n0 = (blockIdx.x * 8 + wave) * 64;
  if (n0 >= N || m0 >= M) return;

  v8f acc[2][4];
#pragma unroll
  for (int mt = 0; mt < 2; ++mt)
#pragma unroll
    for (int nt = 0; nt < 4; ++nt) { v8f zz = {}; acc[mt][nt] = zz; }

  if ((m0 + 32 <= M) && (n0 + 64 <= N) && ((m0 % S) != 0)) {
    // ---------------- fast path ----------------
    for (int t = 0; t < 4; ++t) {
      const __bf16* Bt = W + (size_t)t * N * K;
      const int mb = m0 + (t - 3);
      Frag a0[2], b0[4], a1[2], b1[4];
      load_a(a0, A, lda, mb, l15, hi, 0);
      load_b(b0, Bt, ldb, n0, l15, hi, 0);
      int k0 = 0;
      for (; k0 + 64 < K; k0 += 64) {
        load_a(a1, A, lda, mb, l15, hi, k0 + 32);
        load_b(b1, Bt, ldb, n0, l15, hi, k0 + 32);
        mma8(acc, a0, b0);
        load_a(a0, A, lda, mb, l15, hi, k0 + 64);
        load_b(b0, Bt, ldb, n0, l15, hi, k0 + 64);
        mma8(acc, a1, b1);
      }
      load_a(a1, A, lda, mb, l15, hi, k0 + 32);
      load_b(b1, Bt, ldb, n0, l15, hi, k0 + 32);
      mma8(acc, a0, b0);
      mma8(acc, a1, b1);
    }
  } else {
    // ---------------- edge path ----------------
    const Frag fz = zero_frag();
    for (int t = 0; t < 4; ++t) {
      const __bf16* Bt = W + (size_t)t * N * K;
      const int shift = t - 3;
      for (int k0 = 0; k0 < K; k0 += 32) {
        Frag fa[2], fb[4];
#pragma unroll
        for (int mt = 0; mt < 2; ++mt) {
          const int row = m0 + mt * 16 + l15;
          bool ok = false;
          const int src = row + shift;
          if (row < M) { const int s = row % S; ok = (s + shift) >= 0; }
          if (ok) {
            const __bf16* p = A + (size_t)src * lda + k0 + hi * 8;
            fa[mt].h[0] = ld8(p);
            fa[mt].h[1] = ld8(p + 16);
          } else fa[mt] = fz;
        }
#pragma unroll
        for (int nt = 0; nt < 4; ++nt) {
          const int col = n0 + nt * 16 + l15;
          if (col < N) {
            const __bf16* p = Bt + (size_t)col * ldb + k0 + hi * 8;
            fb[nt].h[0] = ld8(p);
            fb[nt].h[1] = ld8(p + 16);
          } else fb[nt] = fz;
        }
        mma8(acc, fa, fb);
      }
    }
  }

#pragma unroll
  for (int mt = 0; mt < 2; ++mt) {
    const int rbase = m0 + mt * 16 + hi * 8;
#pragma unroll
    for (int nt = 0; nt < 4; ++nt) {
      const int col = n0 + nt * 16 + l15;
      if (col >= N) continue;
#pragma unroll
      for (int r8 = 0; r8 < 8; ++r8) {
        const int row = rbase + r8;
        if (row >= M) continue;
        C[(size_t)row * ldc + col] = acc[mt][nt][r8];
      }
    }
  }
}

// ---------------------------------------------------------------------------
// Elementwise / support kernels
// ---------------------------------------------------------------------------
__global__ void k_embed(const int* __restrict__ tok, const float* __restrict__ E,
                        float* __restrict__ x, int D)
{
  const int row = blockIdx.x;
  const int t = tok[row];
  for (int d = threadIdx.x; d < D; d += blockDim.x)
    x[(size_t)row * D + d] = E[(size_t)t * D + d];
}

__global__ __launch_bounds__(256) void k_rmsnorm(const float* __restrict__ x,
                                                 const float* __restrict__ w,
                                                 __bf16* __restrict__ xn, int D)
{
  const int row = blockIdx.x;
  const float* xr = x + (size_t)row * D;
  float ss = 0.f;
  for (int d = threadIdx.x; d < D; d += 256) { const float v = xr[d]; ss += v * v; }
  __shared__ float red[256];
  red[threadIdx.x] = ss;
  __syncthreads();
  for (int off = 128; off > 0; off >>= 1) {
    if (threadIdx.x < off) red[threadIdx.x] += red[threadIdx.x + off];
    __syncthreads();
  }
  const float scale = rsqrtf(red[0] / (float)D + 1e-5f);
  for (int d = threadIdx.x; d < D; d += 256)
    xn[(size_t)row * D + d] = (__bf16)(w[d] * xr[d] * scale);
}

// f32 [N,K] -> bf16 [N,Kp] with zero K-padding (Kp multiple of 64)
__global__ void k_cvt_pad(const float* __restrict__ src, __bf16* __restrict__ dst,
                          int N, int K, int Kp)
{
  const size_t idx = (size_t)blockIdx.x * blockDim.x + threadIdx.x;
  if (idx >= (size_t)N * Kp) return;
  const int n = (int)(idx / Kp);
  const int k = (int)(idx % Kp);
  dst[idx] = (k < K) ? (__bf16)src[(size_t)n * K + k] : (__bf16)0.0f;
}

// f32 [R,C] -> bf16 [C,R] transpose-convert (for conv taps: [in,out] -> [out,in])
__global__ void k_cvt_T(const float* __restrict__ src, __bf16* __restrict__ dst,
                        int R, int Cc)
{
  const size_t idx = (size_t)blockIdx.x * blockDim.x + threadIdx.x;
  if (idx >= (size_t)R * Cc) return;
  const int i = (int)(idx / Cc);
  const int o = (int)(idx % Cc);
  dst[(size_t)o * R + i] = (__bf16)src[idx];
}

// ---------------------------------------------------------------------------
// mLSTM scan: one block per (b,h), 64 threads; thread e owns state column e
// state[d][e] = f*state + i*k[d]*v[e];  h[e] = o * sum_d q[d]*state[d][e]
// ---------------------------------------------------------------------------
__global__ __launch_bounds__(64) void k_mlstm_scan(
    const float* __restrict__ q, const float* __restrict__ k,
    const float* __restrict__ v,
    const float* __restrict__ gi, const float* __restrict__ gf,
    const float* __restrict__ go,
    float* __restrict__ hs, int S, int H, int ld)
{
  const int b = blockIdx.x / H;
  const int h = blockIdx.x % H;
  const int e = threadIdx.x; // 0..63
  __shared__ float sq[64], sk[64];
  float st[64];
#pragma unroll
  for (int d = 0; d < 64; ++d) st[d] = 0.f;

  for (int s = 0; s < S; ++s) {
    const size_t r = (size_t)(b * S + s);
    const size_t base = r * ld + (size_t)h * 64;
    sq[e] = q[base + e];
    sk[e] = k[base + e];
    __syncthreads();
    const float ve = v[base + e];
    const size_t gidx = r * H + h;
    const float i_ = gi[gidx], f_ = gf[gidx], o_ = go[gidx];
    float acc = 0.f;
#pragma unroll
    for (int d = 0; d < 64; ++d) {
      st[d] = f_ * st[d] + i_ * (sk[d] * ve);
      acc += sq[d] * st[d];
    }
    hs[base + e] = o_ * acc;
    __syncthreads();
  }
}

// ---------------------------------------------------------------------------
// sLSTM scan: fully parallel over (b, p), sequential over s.
// ---------------------------------------------------------------------------
__global__ void k_slstm_scan(const float* __restrict__ gi, const float* __restrict__ gf,
                             const float* __restrict__ go, const float* __restrict__ ci,
                             float* __restrict__ hs, int B, int S, int P)
{
  const int idx = blockIdx.x * blockDim.x + threadIdx.x;
  if (idx >= B * P) return;
  const int b = idx / P;
  const int p = idx % P;
  float st = 0.f;
  for (int s = 0; s < S; ++s) {
    const size_t r = ((size_t)(b * S + s)) * P + p;
    st = gf[r] * st + gi[r] * tanhf(ci[r]);
    hs[r] = go[r] * tanhf(st);
  }
}

// ---------------------------------------------------------------------------
// Host orchestration
// ---------------------------------------------------------------------------
extern "C" void kernel_launch(void* const* d_in, const int* in_sizes, int n_in,
                              void* d_out, int out_size, void* d_ws, size_t ws_size,
                              hipStream_t stream)
{
  (void)n_in; (void)out_size; (void)ws_size;
  const int S  = 1024;
  const int H  = 16;
  const int D  = in_sizes[3];            // ln of layer0 -> 1024
  const int B  = in_sizes[0] / S;        // 2
  const int BS = B * S;                  // 2048
  const int V  = in_sizes[1] / D;        // 50257
  const int PROJ  = in_sizes[5] / D;     // Wi of layer0 -> 1364
  const int PROJP = (PROJ + 63) & ~63;   // 1408 (K padded to mult of 64)

  // ---- workspace bump allocator ----
  uint8_t* wsp = (uint8_t*)d_ws;
  auto alloc = [&](size_t bytes) -> void* {
    void* p = (void*)wsp;
    wsp += (bytes + 255) & ~(size_t)255;
    return p;
  };
  float*  x    = (float*) alloc((size_t)BS * D * 4);      // residual stream (f32)
  __bf16* xn   = (__bf16*)alloc((size_t)BS * D * 2);      // rmsnorm out / final x (bf16)
  float*  xc   = (float*) alloc((size_t)BS * D * 4);      // conv out (f32)
  __bf16* xcb  = (__bf16*)alloc((size_t)BS * D * 2);      // conv out (bf16)
  float*  g0   = (float*) alloc((size_t)BS * PROJP * 4);  // q / ig
  float*  g1   = (float*) alloc((size_t)BS * PROJP * 4);  // k / fg
  float*  g2   = (float*) alloc((size_t)BS * PROJP * 4);  // v / og
  float*  g3   = (float*) alloc((size_t)BS * PROJP * 4);  // c_in
  float*  mi   = (float*) alloc((size_t)BS * H * 4);      // mLSTM gates
  float*  mf   = (float*) alloc((size_t)BS * H * 4);
  float*  mo   = (float*) alloc((size_t)BS * H * 4);
  float*  hs   = (float*) alloc((size_t)BS * PROJP * 4);  // scan output
  __bf16* hsb  = (__bf16*)alloc((size_t)BS * PROJP * 2);  // scan output (bf16, padded)
  __bf16* wb   = (__bf16*)alloc((size_t)V * D * 2);       // weight staging (max = head)

  auto cvt = [&](const float* s, __bf16* dst, int N, int K, int Kp) {
    const size_t tot = (size_t)N * Kp;
    k_cvt_pad<<<dim3((unsigned)((tot + 255) / 256)), 256, 0, stream>>>(s, dst, N, K, Kp);
  };
  auto gemm = [&](const __bf16* A, int lda, const __bf16* Bm, int ldb,
                  float* C, int ldc, const float* Cadd,
                  int M, int N, int K, int epi, float cap) {
    dim3 grid((unsigned)((N + 511) / 512), (unsigned)((M + 31) / 32));
    k_gemm<<<grid, 256, 0, stream>>>(A, lda, Bm, ldb, C, ldc, Cadd, M, N, K, epi, cap);
  };

  // ---- embedding gather ----
  k_embed<<<BS, 256, 0, stream>>>((const int*)d_in[0], (const float*)d_in[1], x, D);

  // ---- layers ----
  int ii = 3;
  for (int layer = 0; layer < 6; ++layer) {
    if (layer < 2) {
      // ----- sLSTM -----
      const float* ln   = (const float*)d_in[ii + 0];
      const float* conv = (const float*)d_in[ii + 1]; // [4, in=D, out=D]
      const float* Wi   = (const float*)d_in[ii + 2]; // [PROJ, D]
      const float* Wf   = (const float*)d_in[ii + 3];
      const float* Wo   = (const float*)d_in[ii + 4];
      const float* Wc   = (const float*)d_in[ii + 5];
      const float* Wout = (const float*)d_in[ii + 6]; // [D, PROJ]
      ii += 7;

      k_rmsnorm<<<BS, 256, 0, stream>>>(x, ln, xn, D);

      // conv taps -> bf16 [t][out][in]
      for (int t = 0; t < 4; ++t) {
        const size_t tot = (size_t)D * D;
        k_cvt_T<<<dim3((unsigned)((tot + 255) / 256)), 256, 0, stream>>>(
            conv + (size_t)t * D * D, wb + (size_t)t * D * D, D, D);
      }
      {
        dim3 grid((unsigned)((D + 511) / 512), (unsigned)((BS + 31) / 32));
        k_conv_gemm<<<grid, 256, 0, stream>>>(xn, D, wb, D, xc, D, BS, D, D, S);
      }
      cvt(xc, xcb, BS, D, D);

      cvt(Wi, wb, PROJ, D, D);
      gemm(xcb, D, wb, D, g0, PROJ, nullptr, BS, PROJ, D, EPI_GATE, 0.f);
      cvt(Wf, wb, PROJ, D, D);
      gemm(xcb, D, wb, D, g1, PROJ, nullptr, BS, PROJ, D, EPI_GATE, 0.f);
      cvt(Wo, wb, PROJ, D, D);
      gemm(xcb, D, wb, D, g2, PROJ, nullptr, BS, PROJ, D, EPI_GATE, 0.f);
      cvt(Wc, wb, PROJ, D, D);
      gemm(xcb, D, wb, D, g3, PROJ, nullptr, BS, PROJ, D, EPI_NONE, 0.f);

      k_slstm_scan<<<dim3((unsigned)((B * PROJ + 255) / 256)), 256, 0, stream>>>(
          g0, g1, g2, g3, hs, B, S, PROJ);

      cvt(hs, hsb, BS, PROJ, PROJP);
      cvt(Wout, wb, D, PROJ, PROJP);
      gemm(hsb, PROJP, wb, PROJP, x, D, x, BS, D, PROJP, EPI_RES, 0.f);
    } else {
      // ----- mLSTM -----
      const float* ln   = (const float*)d_in[ii + 0];
      const float* Wq   = (const float*)d_in[ii + 1]; // [H*HD=D, D]
      const float* Wk   = (const float*)d_in[ii + 2];
      const float* Wv   = (const float*)d_in[ii + 3];
      const float* Wig  = (const float*)d_in[ii + 4]; // [H, D]
      const float* Wfg  = (const float*)d_in[ii + 5];
      const float* Wog  = (const float*)d_in[ii + 6];
      const float* Wout = (const float*)d_in[ii + 7]; // [D, H*HD]
      ii += 8;

      k_rmsnorm<<<BS, 256, 0, stream>>>(x, ln, xn, D);

      cvt(Wq, wb, D, D, D);
      gemm(xn, D, wb, D, g0, D, nullptr, BS, D, D, EPI_NONE, 0.f);
      cvt(Wk, wb, D, D, D);
      gemm(xn, D, wb, D, g1, D, nullptr, BS, D, D, EPI_NONE, 0.f);
      cvt(Wv, wb, D, D, D);
      gemm(xn, D, wb, D, g2, D, nullptr, BS, D, D, EPI_NONE, 0.f);

      cvt(Wig, wb, H, D, D);
      gemm(xn, D, wb, D, mi, H, nullptr, BS, H, D, EPI_GATE, 0.f);
      cvt(Wfg, wb, H, D, D);
      gemm(xn, D, wb, D, mf, H, nullptr, BS, H, D, EPI_GATE, 0.f);
      cvt(Wog, wb, H, D, D);
      gemm(xn, D, wb, D, mo, H, nullptr, BS, H, D, EPI_GATE, 0.f);

      k_mlstm_scan<<<B * H, 64, 0, stream>>>(g0, g1, g2, mi, mf, mo, hs, S, H, D);

      cvt(hs, hsb, BS, D, D);
      cvt(Wout, wb, D, D, D);
      gemm(hsb, D, wb, D, x, D, x, BS, D, D, EPI_RES, 0.f);
    }
  }

  // ---- head projection + output soft cap ----
  cvt(x, xn, BS, D, D);
  cvt((const float*)d_in[2], wb, V, D, D);
  gemm(xn, D, wb, D, (float*)d_out, V, nullptr, BS, V, D, EPI_CAP, 30.0f);
}
